// MambaSSM_48730698940563
// MI455X (gfx1250) — matrix operand order repaired and checked by reading, not compiled, not run on hardware
//
#include <hip/hip_runtime.h>

// Problem constants (from the reference)
#define D_MODEL 128
#define D_INNER 256
#define D_STATE 256
#define SEQ_L   2048
#define BATCH   2
#define NTOK    (BATCH * SEQ_L)          // 4096 tokens

typedef __attribute__((ext_vector_type(16))) _Float16 v16h;
typedef __attribute__((ext_vector_type(8)))  float    v8f;

// ---------------------------------------------------------------------------
// WMMA fragment gathers, per CDNA5 ISA 7.12.2 16-bit layouts (wave32).
// A (16x32 f16): lane = hi*16 + m ; element e=(v,s): K = (v>>2)*16 + hi*8 + 2*(v&3) + s
// B (32x16 f16): lane = hi*16 + n ; element e=(v,s): K = (v>>2)*16 + hi*8 + 2*(v&3) + s
// LDS tiles are stored row-per-"lane-index": A as [m][k], B transposed as [n][k],
// padded to stride 40 halves to avoid bank conflicts.
// ---------------------------------------------------------------------------
__device__ __forceinline__ void load_frag_a(const _Float16* As, int tm, int lane, v16h& a) {
  const int hi = lane >> 4, mr = lane & 15;
  const _Float16* row = As + (tm + mr) * 40;
#pragma unroll
  for (int v = 0; v < 8; ++v) {
    const int k = ((v >> 2) << 4) + (hi << 3) + ((v & 3) << 1);
    a[2 * v]     = row[k];
    a[2 * v + 1] = row[k + 1];
  }
}

__device__ __forceinline__ void load_frag_b(const _Float16* Bs, int tn, int lane, v16h& b) {
  const int hi = lane >> 4, nr = lane & 15;
  const _Float16* row = Bs + (tn + nr) * 40;
#pragma unroll
  for (int v = 0; v < 8; ++v) {
    const int k = ((v >> 2) << 4) + (hi << 3) + ((v & 3) << 1);
    b[2 * v]     = row[k];
    b[2 * v + 1] = row[k + 1];
  }
}

// 256-thread block = 8 waves, macrotile 32(M) x 64(N), K stepped by 32.
// Each wave owns one 16x16 output tile. A,B are f32 in memory, converted to
// f16 on the way into LDS; accumulation stays f32 in the WMMA accumulator.
__device__ __forceinline__ v8f gemm_core(const float* __restrict__ Ag, int lda,
                                         const float* __restrict__ Bg, int ldb,
                                         int ktot, int rowA0, int colB0,
                                         _Float16* As, _Float16* Bs)
{
  const int tid  = threadIdx.x;
  const int lane = tid & 31;
  const int w    = tid >> 5;
  const int wm   = w >> 2;   // 0..1 : which 16 rows
  const int wn   = w & 3;    // 0..3 : which 16 cols
  v8f acc = {};
  for (int k0 = 0; k0 < ktot; k0 += 32) {
    __syncthreads();
    // Stage A: 32 rows x 32 k, one float4 per thread
    {
      const int r = tid >> 3;
      const int c = (tid & 7) << 2;
      const float4 v = *reinterpret_cast<const float4*>(Ag + (size_t)(rowA0 + r) * lda + k0 + c);
      _Float16* dst = As + r * 40 + c;
      dst[0] = (_Float16)v.x; dst[1] = (_Float16)v.y;
      dst[2] = (_Float16)v.z; dst[3] = (_Float16)v.w;
    }
    // Stage B: 32 k x 64 n, stored transposed [n][k], two float4 per thread
    {
      const int r = tid >> 3;          // k row 0..31
      const int c = (tid & 7) << 3;    // n col 0,8,..,56
      const float* src = Bg + (size_t)(k0 + r) * ldb + colB0 + c;
      const float4 v0 = *reinterpret_cast<const float4*>(src);
      const float4 v1 = *reinterpret_cast<const float4*>(src + 4);
      Bs[(c + 0) * 40 + r] = (_Float16)v0.x;
      Bs[(c + 1) * 40 + r] = (_Float16)v0.y;
      Bs[(c + 2) * 40 + r] = (_Float16)v0.z;
      Bs[(c + 3) * 40 + r] = (_Float16)v0.w;
      Bs[(c + 4) * 40 + r] = (_Float16)v1.x;
      Bs[(c + 5) * 40 + r] = (_Float16)v1.y;
      Bs[(c + 6) * 40 + r] = (_Float16)v1.z;
      Bs[(c + 7) * 40 + r] = (_Float16)v1.w;
    }
    if (k0 + 32 < ktot) {  // pull next K tile toward the WGP (global_prefetch_b8)
      __builtin_prefetch(Ag + (size_t)(rowA0 + (tid >> 3)) * lda + k0 + 32, 0, 1);
      __builtin_prefetch(Bg + (size_t)(k0 + 32 + (tid >> 3)) * ldb + colB0, 0, 1);
    }
    __syncthreads();
    v16h a, b;
    load_frag_a(As, wm * 16, lane, a);
    load_frag_b(Bs, wn * 16, lane, b);
    acc = __builtin_amdgcn_wmma_f32_16x16x32_f16(false, a, false, b, (short)0, acc,
                                                 false, false);
  }
  return acc;
}

// ---------------------------------------------------------------------------
// Kernel 0: prep. Block 256 = column-sum of second half of W_x (the only thing
// the reference ever uses of x_dbl). Blocks 0..255 = transpose A = -exp(A_log)
// into state-major layout so the scan's loads are lane-coalesced.
// ---------------------------------------------------------------------------
__global__ void __launch_bounds__(256) k_prep(const float* __restrict__ Wx,
                                              const float* __restrict__ Alog,
                                              float* __restrict__ wsum,
                                              float* __restrict__ At)
{
  const int tid = threadIdx.x;
  if (blockIdx.x == 256) {
    float s = 0.f;
    for (int j = 0; j < D_STATE; ++j)
      s += Wx[(size_t)tid * (2 * D_STATE) + D_STATE + j];
    wsum[tid] = s;
  } else {
    const int s = blockIdx.x;   // state index
    const int i = tid;          // inner index
    At[(size_t)s * D_INNER + i] = -__expf(Alog[(size_t)i * D_STATE + s]);
  }
}

// ---------------------------------------------------------------------------
// Kernel 1: xz = x @ W_in  (4096x128 @ 128x512), split into xs | z
// ---------------------------------------------------------------------------
__global__ void __launch_bounds__(256) k_gemm_in(const float* __restrict__ x,
                                                 const float* __restrict__ Win,
                                                 float* __restrict__ xs,
                                                 float* __restrict__ zbuf)
{
  __shared__ _Float16 As[32 * 40];
  __shared__ _Float16 Bs[64 * 40];
  const int rowA0 = blockIdx.y * 32;
  const int colB0 = blockIdx.x * 64;
  v8f acc = gemm_core(x, D_MODEL, Win, 2 * D_INNER, D_MODEL, rowA0, colB0, As, Bs);
  const int lane = threadIdx.x & 31, w = threadIdx.x >> 5;
  const int hi = lane >> 4, mr = lane & 15;
  const int row0 = rowA0 + (w >> 2) * 16 + hi * 8;
  const int col  = colB0 + (w & 3) * 16 + mr;
#pragma unroll
  for (int r = 0; r < 8; ++r) {
    const int row = row0 + r;
    const float v = acc[r];
    if (col < D_INNER) xs[(size_t)row * D_INNER + col] = v;
    else               zbuf[(size_t)row * D_INNER + (col - D_INNER)] = v;
  }
}

// ---------------------------------------------------------------------------
// Kernel 2: causal depthwise conv (k=4) + S[tok] = xc . wsum (block reduce)
// One token per block, channel = threadIdx.x (coalesced).
// ---------------------------------------------------------------------------
__global__ void __launch_bounds__(256) k_conv(const float* __restrict__ xs,
                                              const float* __restrict__ cw,
                                              const float* __restrict__ cb,
                                              const float* __restrict__ wsum,
                                              float* __restrict__ xc,
                                              float* __restrict__ S)
{
  const int tok = blockIdx.x;
  const int c   = threadIdx.x;
  const int l   = tok & (SEQ_L - 1);
  float acc = cb[c];
#pragma unroll
  for (int t = 0; t < 4; ++t) {
    const int ll = l - 3 + t;
    if (ll >= 0) acc += xs[(size_t)(tok - 3 + t) * D_INNER + c] * cw[c * 4 + t];
  }
  xc[(size_t)tok * D_INNER + c] = acc;

  __shared__ float red[256];
  red[c] = acc * wsum[c];
  __syncthreads();
  for (int off = 128; off > 0; off >>= 1) {
    if (c < off) red[c] += red[c + off];
    __syncthreads();
  }
  if (c == 0) S[tok] = red[0];
}

// ---------------------------------------------------------------------------
// Kernel 3: delta = softplus(xc @ W_dt + b_dt); fused epilogue also emits
// bv = delta * xc * S (the scan's additive term).
// ---------------------------------------------------------------------------
__global__ void __launch_bounds__(256) k_gemm_dt(const float* __restrict__ xc,
                                                 const float* __restrict__ Wdt,
                                                 const float* __restrict__ bdt,
                                                 const float* __restrict__ S,
                                                 float* __restrict__ dlt,
                                                 float* __restrict__ bv)
{
  __shared__ _Float16 As[32 * 40];
  __shared__ _Float16 Bs[64 * 40];
  const int rowA0 = blockIdx.y * 32;
  const int colB0 = blockIdx.x * 64;
  v8f acc = gemm_core(xc, D_INNER, Wdt, D_STATE, D_INNER, rowA0, colB0, As, Bs);
  const int lane = threadIdx.x & 31, w = threadIdx.x >> 5;
  const int hi = lane >> 4, mr = lane & 15;
  const int row0 = rowA0 + (w >> 2) * 16 + hi * 8;
  const int col  = colB0 + (w & 3) * 16 + mr;
#pragma unroll
  for (int r = 0; r < 8; ++r) {
    const int row = row0 + r;
    const size_t idx = (size_t)row * D_STATE + col;
    const float v  = acc[r] + bdt[col];
    const float sp = fmaxf(v, 0.f) + log1pf(__expf(-fabsf(v)));  // stable softplus
    dlt[idx] = sp;
    bv[idx]  = sp * xc[idx] * S[row];
  }
}

// ---------------------------------------------------------------------------
// Kernel 4: the sequential scan. y_t = M_t y_{t-1} + b_t with
// M_t[i,s] = exp(delta_t[i] * A[i,s]) generated on the fly (never stored).
// One persistent 256-thread workgroup per batch; y lives in LDS and is
// broadcast-read; A is pre-transposed so each v_exp feed is lane-coalesced.
// Output is fused with the gate: A3 = y * silu(z).
// ---------------------------------------------------------------------------
__global__ void __launch_bounds__(256) k_scan(const float* __restrict__ dlt,
                                              const float* __restrict__ bv,
                                              const float* __restrict__ At,
                                              const float* __restrict__ zbuf,
                                              float* __restrict__ A3)
{
  const int b = blockIdx.x;
  const int i = threadIdx.x;
  __shared__ __attribute__((aligned(16))) float y[D_INNER];
  y[i] = 0.f;
  __syncthreads();
  const float4* y4 = reinterpret_cast<const float4*>(y);
  for (int t = 0; t < SEQ_L; ++t) {
    const size_t base = ((size_t)b * SEQ_L + t) * D_INNER;
    const float d = dlt[base + i];
    float acc = 0.f;
#pragma unroll 2
    for (int s = 0; s < D_STATE; s += 4) {
      const float4 yv = y4[s >> 2];                        // LDS broadcast
      const float a0 = At[(size_t)(s + 0) * D_INNER + i];  // coalesced, L2-hot
      const float a1 = At[(size_t)(s + 1) * D_INNER + i];
      const float a2 = At[(size_t)(s + 2) * D_INNER + i];
      const float a3 = At[(size_t)(s + 3) * D_INNER + i];
      acc += __expf(d * a0) * yv.x + __expf(d * a1) * yv.y
           + __expf(d * a2) * yv.z + __expf(d * a3) * yv.w;
    }
    const float ynew = acc + bv[base + i];
    __syncthreads();
    y[i] = ynew;
    __syncthreads();
    const float zz = zbuf[base + i];
    A3[base + i] = ynew * (zz / (1.f + __expf(-zz)));      // fused y * silu(z)
  }
}

// ---------------------------------------------------------------------------
// Kernel 5: out = A3 @ W_out  (4096x256 @ 256x128), fp32 output
// ---------------------------------------------------------------------------
__global__ void __launch_bounds__(256) k_gemm_out(const float* __restrict__ A3,
                                                  const float* __restrict__ Wout,
                                                  float* __restrict__ out)
{
  __shared__ _Float16 As[32 * 40];
  __shared__ _Float16 Bs[64 * 40];
  const int rowA0 = blockIdx.y * 32;
  const int colB0 = blockIdx.x * 64;
  v8f acc = gemm_core(A3, D_INNER, Wout, D_MODEL, D_INNER, rowA0, colB0, As, Bs);
  const int lane = threadIdx.x & 31, w = threadIdx.x >> 5;
  const int hi = lane >> 4, mr = lane & 15;
  const int row0 = rowA0 + (w >> 2) * 16 + hi * 8;
  const int col  = colB0 + (w & 3) * 16 + mr;
#pragma unroll
  for (int r = 0; r < 8; ++r)
    out[(size_t)(row0 + r) * D_MODEL + col] = acc[r];
}

// ---------------------------------------------------------------------------
extern "C" void kernel_launch(void* const* d_in, const int* in_sizes, int n_in,
                              void* d_out, int out_size, void* d_ws, size_t ws_size,
                              hipStream_t stream)
{
  (void)in_sizes; (void)n_in; (void)out_size; (void)ws_size;
  const float* x     = (const float*)d_in[0];
  const float* Win   = (const float*)d_in[1];
  const float* convw = (const float*)d_in[2];
  const float* convb = (const float*)d_in[3];
  const float* Wx    = (const float*)d_in[4];
  const float* Wdt   = (const float*)d_in[5];
  const float* bdt   = (const float*)d_in[6];
  const float* Alog  = (const float*)d_in[7];
  const float* Wout  = (const float*)d_in[8];
  float* out = (float*)d_out;

  // workspace layout (floats): ~25.4 MB total
  float* ws = (float*)d_ws;
  const size_t TOKD = (size_t)NTOK * D_INNER;   // 1,048,576
  float* xs   = ws;
  float* zbuf = xs   + TOKD;
  float* xc   = zbuf + TOKD;
  float* dlt  = xc   + TOKD;
  float* bv   = dlt  + TOKD;
  float* A3   = bv   + TOKD;
  float* S    = A3   + TOKD;          // NTOK
  float* wsum = S    + NTOK;          // D_INNER
  float* At   = wsum + D_INNER;       // D_STATE * D_INNER

  k_prep    <<<dim3(257),                            dim3(256), 0, stream>>>(Wx, Alog, wsum, At);
  k_gemm_in <<<dim3((2 * D_INNER) / 64, NTOK / 32),  dim3(256), 0, stream>>>(x, Win, xs, zbuf);
  k_conv    <<<dim3(NTOK),                           dim3(256), 0, stream>>>(xs, convw, convb, wsum, xc, S);
  k_gemm_dt <<<dim3(D_STATE / 64, NTOK / 32),        dim3(256), 0, stream>>>(xc, Wdt, bdt, S, dlt, bv);
  k_scan    <<<dim3(BATCH),                          dim3(256), 0, stream>>>(dlt, bv, At, zbuf, A3);
  k_gemm_out<<<dim3(D_MODEL / 64, NTOK / 32),        dim3(256), 0, stream>>>(A3, Wout, out);
}
